// JANet_56882546868850
// MI455X (gfx1250) — matrix operand order repaired and checked by reading, not compiled, not run on hardware
//
#include <hip/hip_runtime.h>

// JANet: h_final from 1024-step gated recurrence, then dense head.
// prep  : swizzle weights into WMMA fragment layout (hw bf16 converts), zero h0/barrier
// proj  : WMMA bf16 GEMM xz = x@kernel + bias, stored [T,B,2C] fp32
// scan  : persistent 16-block kernel, 1024 steps, atomic grid barrier;
//         R fragments LDS-resident (loaded once), cell state in registers,
//         h double-buffered through L2 in A-frag layout
// dense : tiny head GEMV

typedef __attribute__((ext_vector_type(16))) __bf16 v16bf;
typedef __attribute__((ext_vector_type(16))) unsigned short v16us;
typedef __attribute__((ext_vector_type(8))) float v8f;

union frag_u { v16us us; v16bf bf; };

#define T_STEPS 1024
#define BATCH   256
#define IDIM    128
#define CELLS   256
#define NGATE   512   /* 2*CELLS */
#define ODIM    10

// ---- workspace layout (bytes) ----
#define XZ_OFF   0ull
#define XZ_BYTES (134217728ull * 4ull)            /* T*B*2C fp32 = 512 MB */
#define RSW_OFF  (XZ_OFF + XZ_BYTES)
#define RSW_BYTES (131072ull * 2ull)              /* 256x512 bf16, B-frag layout */
#define WSW_OFF  (RSW_OFF + RSW_BYTES)
#define WSW_BYTES (65536ull * 2ull)               /* 128x512 bf16, B-frag layout */
#define HB_OFF   (WSW_OFF + WSW_BYTES)
#define HB_BYTES (2ull * 65536ull * 2ull)         /* double-buffered h, A-frag layout */
#define BAR_OFF  (HB_OFF + HB_BYTES)

// hardware round-to-nearest-even bf16 convert (v_cvt_*bf16*)
__device__ inline unsigned short f2bf(float f) {
    __bf16 b = (__bf16)f;
    unsigned short u; __builtin_memcpy(&u, &b, 2);
    return u;
}
__device__ inline float bf2f(unsigned short h) {
    unsigned u = ((unsigned)h) << 16;
    float f; __builtin_memcpy(&f, &u, 4);
    return f;
}

// A-matrix 16x32 bf16 fragment: position of element (m 0..15, k 0..31)
// per ISA 7.12.2: lane = m + 16*((k%16)/8), K = (v%4)*2 + (v/4)*16 + half*8
__device__ inline int afrag_pos(int m, int k) {
    int half = (k & 15) >> 3;
    int lane = (m & 15) + 16 * half;
    int rem2 = k & 7;
    int v = (rem2 >> 1) + 4 * (k >> 4);
    return lane * 16 + 2 * v + (k & 1);
}
// B-matrix 32x16 bf16 fragment: element (n 0..15, k 0..31)
__device__ inline int bfrag_pos(int n, int k) {
    int half = (k & 15) >> 3;
    int lane = (n & 15) + 16 * half;
    int v = ((k & 7) >> 1) + 4 * (k >> 4);
    return lane * 16 + 2 * v + (k & 1);
}

// ---------------- prep: swizzle weights, zero h0 + barrier ----------------
__global__ void janet_prep(const float* __restrict__ Wk, const float* __restrict__ Rk,
                           unsigned short* __restrict__ Wsw, unsigned short* __restrict__ Rsw,
                           unsigned short* __restrict__ hbuf, unsigned* __restrict__ bar) {
    int id = blockIdx.x * blockDim.x + threadIdx.x;
    int stride = gridDim.x * blockDim.x;
    for (int i = id; i < IDIM * NGATE; i += stride) {         // kernel [128,512]
        int k = i >> 9, n = i & 511;
        int idx = ((k >> 5) * 32 + (n >> 4)) * 512 + bfrag_pos(n & 15, k & 31);
        Wsw[idx] = f2bf(Wk[i]);
    }
    for (int i = id; i < CELLS * NGATE; i += stride) {        // recurrent [256,512]
        int k = i >> 9, n = i & 511;
        int idx = ((k >> 5) * 32 + (n >> 4)) * 512 + bfrag_pos(n & 15, k & 31);
        Rsw[idx] = f2bf(Rk[i]);
    }
    for (int i = id; i < 2 * BATCH * CELLS; i += stride) hbuf[i] = 0;
    if (id == 0) *bar = 0u;
}

// ---------------- proj: xz[t*B+b, n] = (x @ kernel)[.,n] + bias[n] ----------------
// M = B*T (row m' = t*B + b so the [T,B,2C] store is coalesced), N = 512, K = 128.
__global__ void __launch_bounds__(256)
janet_proj(const float* __restrict__ x, const unsigned short* __restrict__ Wsw,
           const float* __restrict__ bias, float* __restrict__ xz) {
    const int tid = threadIdx.x, lane = tid & 31, w = tid >> 5;
    const int nBlk = blockIdx.x & 7;        // 8 col-blocks of 64
    const int mBlk = blockIdx.x >> 3;       // 2048 row-blocks of 128
    const int mBase = mBlk * 128 + w * 16;
    const int mrow = mBase + (lane & 15);   // A-fragment row (m' = t*B + b)
    const int b = mrow & 255, t = mrow >> 8;
    const float* xrow = x + ((size_t)b * T_STEPS + t) * IDIM;
    const int nTile0 = nBlk * 4;

    v8f acc[4] = {v8f{}, v8f{}, v8f{}, v8f{}};
#pragma unroll
    for (int kk = 0; kk < 4; ++kk) {
        __bf16 abuf[16];
#pragma unroll
        for (int v = 0; v < 8; ++v) {       // gather A in fragment order (hw cvt to bf16)
            int k = kk * 32 + (v & 3) * 2 + (v >> 2) * 16 + (lane >> 4) * 8;
            abuf[2 * v]     = (__bf16)xrow[k];
            abuf[2 * v + 1] = (__bf16)xrow[k + 1];
        }
        frag_u a; __builtin_memcpy(&a.bf, abuf, 32);
        const unsigned short* wb = Wsw + (size_t)(kk * 32) * 512 + lane * 16;
#pragma unroll
        for (int q = 0; q < 4; ++q) {
            frag_u bf; bf.us = *(const v16us*)(wb + (size_t)(nTile0 + q) * 512);
            acc[q] = __builtin_amdgcn_wmma_f32_16x16x32_bf16(
                false, a.bf, false, bf.bf, (short)0, acc[q], false, false);
        }
    }
    const int mEl = mBase + (lane >> 4) * 8;
    const int nEl = nBlk * 64 + (lane & 15);
#pragma unroll
    for (int q = 0; q < 4; ++q)
#pragma unroll
        for (int g = 0; g < 8; ++g) {
            int m = mEl + g, n = nEl + q * 16;
            xz[(size_t)m * NGATE + n] = acc[q][g] + bias[n];
        }
}

// ---------------- scan: 1024 sequential steps, persistent 16 blocks ----------------
// Block = 64 batch-rows x 64 cells; wave = 16 rows x 32 cells, BOTH gates so the
// c update is thread-local. The block's 64 R fragments (64 KB) are copied to LDS
// once and re-read 1024 times via ds_load_b128; h A-fragments preloaded per step.
__global__ void __launch_bounds__(256, 1)
janet_scan(const float* __restrict__ xz, const unsigned short* __restrict__ Rsw,
           unsigned short* __restrict__ hbuf, unsigned* __restrict__ bar) {
    const int tid = threadIdx.x, lane = tid & 31, w = tid >> 5;
    const int mBlk = blockIdx.x & 3;          // batch tile of 64
    const int nBlk = blockIdx.x >> 2;         // cell tile of 64
    const int mStrip = w & 3, nHalf = w >> 2;
    const int mTileG = mBlk * 4 + mStrip;     // 16-row tile index (0..15)

    // ---- LDS-resident R: frag local index = k8*8 + gate*4 + q (q = col-tile 0..3) ----
    __shared__ unsigned short Rlds[64 * 512];   // 64 KB
    for (int c = tid; c < 64 * 64; c += 256) {  // 4096 x 16-byte chunks
        int f = c >> 6, chunk = c & 63;
        int k8 = f >> 3, rem = f & 7, gate = rem >> 2, qq = rem & 3;
        int gsrc = (k8 * 32 + nBlk * 4 + gate * 16 + qq) * 512 + chunk * 8;
        *(uint4*)(&Rlds[f * 512 + chunk * 8]) = *(const uint4*)(Rsw + gsrc);
    }
    __syncthreads();

    float creg[16];
#pragma unroll
    for (int i = 0; i < 16; ++i) creg[i] = 0.f;

    const int mBase = mBlk * 64 + mStrip * 16 + (lane >> 4) * 8;
    const int jBase = nBlk * 64 + nHalf * 32 + (lane & 15);

    // t-invariant epilogue offsets: xz offsets and swizzled h store indices
    int offF[16], hidx[16];
#pragma unroll
    for (int q = 0; q < 2; ++q)
#pragma unroll
        for (int g = 0; g < 8; ++g) {
            int m = mBase + g, j = jBase + q * 16, ci = q * 8 + g;
            offF[ci] = m * NGATE + j;
            hidx[ci] = ((j >> 5) * 16 + (m >> 4)) * 512 + afrag_pos(m & 15, j & 31);
        }
    const unsigned short* rbase = &Rlds[lane * 16];
    const int lq0 = nHalf * 2;                 // local col-tile for accF/G[0]

    for (int t = 0; t < T_STEPS; ++t) {
        const unsigned short* hc = hbuf + (size_t)(t & 1) * (BATCH * CELLS);
        unsigned short* hn = hbuf + (size_t)((t + 1) & 1) * (BATCH * CELLS);

        // preload all 8 A fragments (overlapping global loads)
        frag_u aall[8];
#pragma unroll
        for (int k8 = 0; k8 < 8; ++k8)
            aall[k8].us = *(const v16us*)(hc + (size_t)(k8 * 16 + mTileG) * 512 + lane * 16);

        v8f accF[2] = {v8f{}, v8f{}};
        v8f accG[2] = {v8f{}, v8f{}};
#pragma unroll
        for (int k8 = 0; k8 < 8; ++k8) {
            frag_u b0, b1, b2, b3;
            b0.us = *(const v16us*)(rbase + (k8 * 8 + 0 + lq0    ) * 512);
            b1.us = *(const v16us*)(rbase + (k8 * 8 + 0 + lq0 + 1) * 512);
            b2.us = *(const v16us*)(rbase + (k8 * 8 + 4 + lq0    ) * 512);  // tanh gate
            b3.us = *(const v16us*)(rbase + (k8 * 8 + 4 + lq0 + 1) * 512);
            accF[0] = __builtin_amdgcn_wmma_f32_16x16x32_bf16(false, aall[k8].bf, false, b0.bf, (short)0, accF[0], false, false);
            accF[1] = __builtin_amdgcn_wmma_f32_16x16x32_bf16(false, aall[k8].bf, false, b1.bf, (short)0, accF[1], false, false);
            accG[0] = __builtin_amdgcn_wmma_f32_16x16x32_bf16(false, aall[k8].bf, false, b2.bf, (short)0, accG[0], false, false);
            accG[1] = __builtin_amdgcn_wmma_f32_16x16x32_bf16(false, aall[k8].bf, false, b3.bf, (short)0, accG[1], false, false);
        }

        const float* xzt = xz + (size_t)t * (BATCH * NGATE);
#pragma unroll
        for (int ci = 0; ci < 16; ++ci) {
            float zf = accF[ci >> 3][ci & 7] + xzt[offF[ci]];
            float zg = accG[ci >> 3][ci & 7] + xzt[offF[ci] + CELLS];
            float f = 1.0f / (1.0f + __expf(-zf));
            float th = 2.0f / (1.0f + __expf(-2.0f * zg)) - 1.0f;   // tanh
            float cn = f * creg[ci] + (1.0f - f) * th;
            creg[ci] = cn;
            hn[hidx[ci]] = f2bf(cn);
        }

        // device-scope barrier across the 16 persistent blocks
        __syncthreads();
        if (tid == 0) {
            __threadfence();
            atomicAdd(bar, 1u);
            unsigned target = 16u * (unsigned)(t + 1);
            while (__hip_atomic_load(bar, __ATOMIC_ACQUIRE, __HIP_MEMORY_SCOPE_AGENT) < target)
                __builtin_amdgcn_s_sleep(1);
        }
        __syncthreads();
    }
}

// ---------------- dense head: out[b,o] = h_final[b,:] @ dense_w + dense_b ----------------
__global__ void janet_dense(const unsigned short* __restrict__ h,
                            const float* __restrict__ Wd, const float* __restrict__ bd,
                            float* __restrict__ out) {
    int id = blockIdx.x * blockDim.x + threadIdx.x;
    if (id >= BATCH * ODIM) return;
    int b = id / ODIM, o = id % ODIM;
    float s = bd[o];
#pragma unroll 4
    for (int j = 0; j < CELLS; ++j) {
        int idx = ((j >> 5) * 16 + (b >> 4)) * 512 + afrag_pos(b & 15, j & 31);
        s += bf2f(h[idx]) * Wd[j * ODIM + o];
    }
    out[id] = s;
}

extern "C" void kernel_launch(void* const* d_in, const int* in_sizes, int n_in,
                              void* d_out, int out_size, void* d_ws, size_t ws_size,
                              hipStream_t stream) {
    const float* x     = (const float*)d_in[0];
    const float* Wk    = (const float*)d_in[1];
    const float* Rk    = (const float*)d_in[2];
    const float* rbias = (const float*)d_in[3];
    const float* Wd    = (const float*)d_in[4];
    const float* bd    = (const float*)d_in[5];
    float* out = (float*)d_out;

    char* ws = (char*)d_ws;
    float*          xz   = (float*)(ws + XZ_OFF);
    unsigned short* Rsw  = (unsigned short*)(ws + RSW_OFF);
    unsigned short* Wsw  = (unsigned short*)(ws + WSW_OFF);
    unsigned short* hbuf = (unsigned short*)(ws + HB_OFF);
    unsigned*       bar  = (unsigned*)(ws + BAR_OFF);

    janet_prep<<<512, 256, 0, stream>>>(Wk, Rk, Wsw, Rsw, hbuf, bar);
    janet_proj<<<16384, 256, 0, stream>>>(x, Wsw, rbias, xz);
    janet_scan<<<16, 256, 0, stream>>>(xz, Rsw, hbuf, bar);
    // T = 1024 is even -> final h lands in buffer 0
    janet_dense<<<(BATCH * ODIM + 255) / 256, 256, 0, stream>>>(hbuf, Wd, bd, out);
}
